// TrackingNNet_75479755259914
// MI455X (gfx1250) — matrix operand order repaired and checked by reading, not compiled, not run on hardware
//
#include <hip/hip_runtime.h>
#include <hip/hip_bf16.h>

// ---------------------------------------------------------------------------
// MI455X (gfx1250) fused GNN pipeline.
// Wave32; matrix ops via V_WMMA_F32_16X16X32_F16 (verified builtin).
// One wave owns a 16-row tile and runs the whole MLP (GEMM + LN + ReLU fused),
// with per-wave LDS staging tiles to convert D-layout -> A-layout between
// layers, ordered with explicit `s_wait_dscnt 0`.
// ---------------------------------------------------------------------------

typedef __attribute__((ext_vector_type(16))) _Float16 v16h;
typedef __attribute__((ext_vector_type(8)))  _Float16 v8h;
typedef __attribute__((ext_vector_type(8)))  float    v8f;

#define LN_EPS 1e-5f

// ---- WMMA operand loaders (per CDNA5 ISA VGPR layouts) ----------
// A (16x32 f16): lanes 0-15 hold row M=lane, K = k0+[0..7] and k0+[16..23];
//                lanes 16-31 hold row M=lane-16, K = k0+[8..15] and k0+[24..31].
__device__ __forceinline__ v16h ldsLoadA(const _Float16* tile, int ldk, int k0, int lane) {
  const int m = lane & 15;
  const _Float16* p = tile + m * ldk + k0 + ((lane & 16) ? 8 : 0);
  v8h lo = *(const v8h*)(p);
  v8h hi = *(const v8h*)(p + 16);
  v16h a;
#pragma unroll
  for (int i = 0; i < 8; ++i) { a[i] = lo[i]; a[i + 8] = hi[i]; }
  return a;
}

// B (32x16 f16), weights stored transposed [N][K]: lanes 0-15 hold col N=lane,
// K = k0+[0..7] and k0+[16..23]; lanes 16-31 hold col N=lane-16, K = k0+[8..15] and k0+[24..31].
__device__ __forceinline__ v16h ldsLoadB(const _Float16* Wt, int K, int n0, int k0, int lane) {
  const int n = n0 + (lane & 15);
  const _Float16* p = Wt + n * K + k0 + ((lane & 16) ? 8 : 0);
  v8h lo = *(const v8h*)(p);
  v8h hi = *(const v8h*)(p + 16);
  v16h b;
#pragma unroll
  for (int i = 0; i < 8; ++i) { b[i] = lo[i]; b[i + 8] = hi[i]; }
  return b;
}

// Wave GEMM: A = [16 x KCH*32] tile in LDS (row-major, ldk), W^T in LDS [64][K].
// Produces 16x64 f32 in 4 accumulator tiles (C/D layout of 16x16 f32).
template <int KCH>
__device__ __forceinline__ void waveGemm(const _Float16* aTile, int ldk,
                                         const _Float16* Wt, int lane, v8f acc[4]) {
  const int K = KCH * 32;
  v16h a[KCH];
#pragma unroll
  for (int kc = 0; kc < KCH; ++kc) a[kc] = ldsLoadA(aTile, ldk, kc * 32, lane);
#pragma unroll
  for (int t = 0; t < 4; ++t) {
    v8f c = {};
#pragma unroll
    for (int kc = 0; kc < KCH; ++kc) {
      v16h b = ldsLoadB(Wt, K, t * 16, kc * 32, lane);
      c = __builtin_amdgcn_wmma_f32_16x16x32_f16(false, a[kc], false, b,
                                                 (short)0, c, false, false);
    }
    acc[t] = c;
  }
}

// Reduce across the 16-lane half-wave (lanes 0-15 / 16-31 independent).
__device__ __forceinline__ float halfReduceSum(float v) {
  v += __shfl_xor(v, 1, 32);
  v += __shfl_xor(v, 2, 32);
  v += __shfl_xor(v, 4, 32);
  v += __shfl_xor(v, 8, 32);
  return v;
}

// Bias + LayerNorm + ReLU fully in registers. Row M = r + 8*(lane>=16) has its
// 64 values at acc[t][r] across 16 lanes (n = (lane&15)+16t).
__device__ __forceinline__ void lnRelu(v8f acc[4], const float* __restrict__ b,
                                       const float* __restrict__ g,
                                       const float* __restrict__ be, int lane) {
  const int nl = lane & 15;
  float bb[4], gg[4], ee[4];
#pragma unroll
  for (int t = 0; t < 4; ++t) {
    bb[t] = b[nl + 16 * t]; gg[t] = g[nl + 16 * t]; ee[t] = be[nl + 16 * t];
  }
#pragma unroll
  for (int t = 0; t < 4; ++t)
#pragma unroll
    for (int r = 0; r < 8; ++r) acc[t][r] += bb[t];
#pragma unroll
  for (int r = 0; r < 8; ++r) {
    float s = acc[0][r] + acc[1][r] + acc[2][r] + acc[3][r];
    float q = acc[0][r] * acc[0][r] + acc[1][r] * acc[1][r] +
              acc[2][r] * acc[2][r] + acc[3][r] * acc[3][r];
    s = halfReduceSum(s);
    q = halfReduceSum(q);
    float mu  = s * (1.f / 64.f);
    float var = q * (1.f / 64.f) - mu * mu;
    float rs  = rsqrtf(var + LN_EPS);
#pragma unroll
    for (int t = 0; t < 4; ++t) {
      float v = (acc[t][r] - mu) * rs * gg[t] + ee[t];
      acc[t][r] = v > 0.f ? v : 0.f;
    }
  }
}

// D-layout accumulators -> LDS row-major [16][64] f16 (stride ld).
__device__ __forceinline__ void stageLds(const v8f acc[4], int lane, _Float16* st, int ld) {
  const int nl  = lane & 15;
  const int mhi = (lane & 16) ? 8 : 0;
#pragma unroll
  for (int t = 0; t < 4; ++t)
#pragma unroll
    for (int r = 0; r < 8; ++r)
      st[(r + mhi) * ld + nl + 16 * t] = (_Float16)acc[t][r];
}

__device__ __forceinline__ void copyH16(_Float16* dst, const _Float16* src, int nHalfs,
                                        int tid, int nthr) {
  const int n4 = nHalfs >> 3;  // 16B chunks
  for (int i = tid; i < n4; i += nthr)
    ((uint4*)dst)[i] = ((const uint4*)src)[i];
}
__device__ __forceinline__ void copyF32(float* dst, const float* src, int n, int tid, int nthr) {
  for (int i = tid; i < n; i += nthr) dst[i] = src[i];
}

// ---------------------------------------------------------------------------
// Weight prep: W [K][N] f32 -> W^T [N][K] f16
// ---------------------------------------------------------------------------
__global__ void transposeW_kernel(const float* __restrict__ W, _Float16* __restrict__ Wt,
                                  int K, int N) {
  int total = K * N;
  for (int idx = blockIdx.x * blockDim.x + threadIdx.x; idx < total;
       idx += gridDim.x * blockDim.x) {
    int n = idx / K, k = idx % K;
    Wt[idx] = (_Float16)W[k * N + n];
  }
}

// ---------------------------------------------------------------------------
// Input MLP: h = ReLU(LN(x @ W + b)), 3 -> 64. One 64-thread block per node.
// ---------------------------------------------------------------------------
__global__ void input_mlp_kernel(const float* __restrict__ x, const float* __restrict__ W,
                                 const float* __restrict__ b, const float* __restrict__ g,
                                 const float* __restrict__ be, _Float16* __restrict__ h,
                                 int Nn) {
  __shared__ float red[64];
  const int v = blockIdx.x;
  const int j = threadIdx.x;
  float o = b[j];
#pragma unroll
  for (int k = 0; k < 3; ++k) o += x[v * 3 + k] * W[k * 64 + j];
  red[j] = o; __syncthreads();
  for (int s = 32; s > 0; s >>= 1) { if (j < s) red[j] += red[j + s]; __syncthreads(); }
  float mu = red[0] * (1.f / 64.f); __syncthreads();
  red[j] = (o - mu) * (o - mu); __syncthreads();
  for (int s = 32; s > 0; s >>= 1) { if (j < s) red[j] += red[j + s]; __syncthreads(); }
  float var = red[0] * (1.f / 64.f);
  float val = (o - mu) * rsqrtf(var + LN_EPS) * g[j] + be[j];
  h[(size_t)v * 64 + j] = (_Float16)(val > 0.f ? val : 0.f);
}

// ---------------------------------------------------------------------------
// Edge MLP (also prune MLP): 128 -> 64 -> 64 -> 64 -> 1, raw logit out.
// ---------------------------------------------------------------------------
struct EdgeParams {
  const _Float16 *Wt0, *Wt1, *Wt2, *w3;
  const float *b0, *g0, *be0, *b1, *g1, *be1, *b2, *g2, *be2, *b3;
};

__global__ void edge_mlp_kernel(const _Float16* __restrict__ h,
                                const int* __restrict__ startI, const int* __restrict__ endI,
                                EdgeParams P, float* __restrict__ out, int E) {
  extern __shared__ char smem[];
  _Float16* sW0 = (_Float16*)smem;        // 64*128
  _Float16* sW1 = sW0 + 64 * 128;         // 64*64
  _Float16* sW2 = sW1 + 64 * 64;          // 64*64
  _Float16* sw3 = sW2 + 64 * 64;          // 64
  float* sPar   = (float*)(sw3 + 64);     // 9*64 + pad
  _Float16* sTiles = (_Float16*)(sPar + 9 * 64 + 8);  // 8 waves * 16*128

  const int tid = threadIdx.x, nthr = blockDim.x;
  copyH16(sW0, P.Wt0, 64 * 128, tid, nthr);
  copyH16(sW1, P.Wt1, 64 * 64, tid, nthr);
  copyH16(sW2, P.Wt2, 64 * 64, tid, nthr);
  copyH16(sw3, P.w3, 64, tid, nthr);
  copyF32(sPar + 0,   P.b0, 64, tid, nthr); copyF32(sPar + 64,  P.g0, 64, tid, nthr);
  copyF32(sPar + 128, P.be0,64, tid, nthr); copyF32(sPar + 192, P.b1, 64, tid, nthr);
  copyF32(sPar + 256, P.g1, 64, tid, nthr); copyF32(sPar + 320, P.be1,64, tid, nthr);
  copyF32(sPar + 384, P.b2, 64, tid, nthr); copyF32(sPar + 448, P.g2, 64, tid, nthr);
  copyF32(sPar + 512, P.be2,64, tid, nthr);
  if (tid == 0) sPar[576] = P.b3[0];
  __syncthreads();

  const int lane = tid & 31, wave = tid >> 5;
  _Float16* sTile = sTiles + wave * (16 * 128);
  const int nTiles = (E + 15) >> 4;
  const int gw = blockIdx.x * (blockDim.x >> 5) + wave;
  const int sw = gridDim.x * (blockDim.x >> 5);
  const int m = lane & 15;

  for (int tile = gw; tile < nTiles; tile += sw) {
    const int e0 = tile * 16;
    // Gather concat(h[start], h[end]) -> [16][128] f16 tile.
    int ei = e0 + m; if (ei >= E) ei = E - 1;
    const int node = (lane & 16) ? endI[ei] : startI[ei];
    const uint4* src = (const uint4*)(h + (size_t)node * 64);
    uint4* dst = (uint4*)(sTile + m * 128 + ((lane & 16) ? 64 : 0));
#pragma unroll
    for (int c = 0; c < 8; ++c) dst[c] = src[c];
    // Prefetch next tile's rows into cache (global_prefetch).
    if (tile + sw < nTiles) {
      int e2 = (tile + sw) * 16 + m; if (e2 >= E) e2 = E - 1;
      const int n2 = (lane & 16) ? endI[e2] : startI[e2];
      __builtin_prefetch(h + (size_t)n2 * 64, 0, 3);
    }
    asm volatile("s_wait_dscnt 0" ::: "memory");

    v8f acc[4];
    waveGemm<4>(sTile, 128, sW0, lane, acc);               // 128 -> 64
    lnRelu(acc, sPar + 0, sPar + 64, sPar + 128, lane);
    stageLds(acc, lane, sTile, 128);
    asm volatile("s_wait_dscnt 0" ::: "memory");

    waveGemm<2>(sTile, 128, sW1, lane, acc);               // 64 -> 64
    lnRelu(acc, sPar + 192, sPar + 256, sPar + 320, lane);
    stageLds(acc, lane, sTile, 128);
    asm volatile("s_wait_dscnt 0" ::: "memory");

    waveGemm<2>(sTile, 128, sW2, lane, acc);               // 64 -> 64
    lnRelu(acc, sPar + 384, sPar + 448, sPar + 512, lane);
    stageLds(acc, lane, sTile, 128);
    asm volatile("s_wait_dscnt 0" ::: "memory");

    // Final 64 -> 1 dot product (split across half-waves).
    float part = 0.f;
    const int joff = (lane & 16) ? 32 : 0;
#pragma unroll
    for (int j = 0; j < 32; ++j)
      part += (float)sTile[m * 128 + joff + j] * (float)sw3[joff + j];
    part += __shfl_xor(part, 16, 32);
    if (lane < 16 && (e0 + m) < E) out[e0 + m] = part + sPar[576];
  }
}

// ---------------------------------------------------------------------------
// Message scatter: mi[end] += sigmoid(e)*h[start]; mo[start] += sigmoid(e)*h[end]
// ---------------------------------------------------------------------------
__global__ void scatter_kernel(const _Float16* __restrict__ h, const int* __restrict__ startI,
                               const int* __restrict__ endI, const float* __restrict__ eraw,
                               float* __restrict__ mi, float* __restrict__ mo, int E) {
  const long total = (long)E * 8;
  const long stride = (long)gridDim.x * blockDim.x;
  for (long w = (long)blockIdx.x * blockDim.x + threadIdx.x; w < total; w += stride) {
    const int e = (int)(w >> 3);
    const int c0 = (int)(w & 7) * 8;
    const float s = 1.f / (1.f + __expf(-eraw[e]));
    const int vs = startI[e], ve = endI[e];
    const _Float16* hs = h + (size_t)vs * 64 + c0;
    const _Float16* he = h + (size_t)ve * 64 + c0;
    float* pmi = mi + (size_t)ve * 64 + c0;
    float* pmo = mo + (size_t)vs * 64 + c0;
#pragma unroll
    for (int j = 0; j < 8; ++j) {
      atomicAdd(pmi + j, s * (float)hs[j]);
      atomicAdd(pmo + j, s * (float)he[j]);
    }
  }
}

// ---------------------------------------------------------------------------
// Node MLP: concat(mi,mo,h) 192 -> 64 -> 64 -> 64 -> 64 (all LN+ReLU) + h.
// ---------------------------------------------------------------------------
struct NodeParams {
  const _Float16 *Wt0, *Wt1, *Wt2, *Wt3;
  const float *b[4], *g[4], *be[4];
};

__global__ void node_mlp_kernel(_Float16* __restrict__ h, const float* __restrict__ mi,
                                const float* __restrict__ mo, NodeParams P, int Nn) {
  extern __shared__ char smem[];
  _Float16* sW0 = (_Float16*)smem;        // 64*192
  _Float16* sW1 = sW0 + 64 * 192;
  _Float16* sW2 = sW1 + 64 * 64;
  _Float16* sW3 = sW2 + 64 * 64;
  float* sPar   = (float*)(sW3 + 64 * 64);            // 12*64
  _Float16* sTiles = (_Float16*)(sPar + 12 * 64);     // 8 waves * 16*192

  const int tid = threadIdx.x, nthr = blockDim.x;
  copyH16(sW0, P.Wt0, 64 * 192, tid, nthr);
  copyH16(sW1, P.Wt1, 64 * 64, tid, nthr);
  copyH16(sW2, P.Wt2, 64 * 64, tid, nthr);
  copyH16(sW3, P.Wt3, 64 * 64, tid, nthr);
#pragma unroll
  for (int l = 0; l < 4; ++l) {
    copyF32(sPar + l * 192 + 0,   P.b[l],  64, tid, nthr);
    copyF32(sPar + l * 192 + 64,  P.g[l],  64, tid, nthr);
    copyF32(sPar + l * 192 + 128, P.be[l], 64, tid, nthr);
  }
  __syncthreads();

  const int lane = tid & 31, wave = tid >> 5;
  _Float16* tileG = sTiles + wave * (16 * 192);
  const int nTiles = (Nn + 15) >> 4;
  const int gw = blockIdx.x * (blockDim.x >> 5) + wave;
  const int sw = gridDim.x * (blockDim.x >> 5);
  const int m = lane & 15;

  for (int tile = gw; tile < nTiles; tile += sw) {
    const int n0 = tile * 16;
    int v = n0 + m; if (v >= Nn) v = Nn - 1;
    // Gather concat(mi, mo, h) -> [16][192] f16; each lane fills 12 8-col chunks.
    const int ccBase = (lane & 16) ? 12 : 0;
    for (int cc = ccBase; cc < ccBase + 12; ++cc) {
      const int c0 = cc * 8;
      _Float16* dst = tileG + m * 192 + c0;
      if (cc < 8) {
        const float* s = mi + (size_t)v * 64 + c0;
#pragma unroll
        for (int j = 0; j < 8; ++j) dst[j] = (_Float16)s[j];
      } else if (cc < 16) {
        const float* s = mo + (size_t)v * 64 + (c0 - 64);
#pragma unroll
        for (int j = 0; j < 8; ++j) dst[j] = (_Float16)s[j];
      } else {
        const _Float16* s = h + (size_t)v * 64 + (c0 - 128);
        *(uint4*)dst = *(const uint4*)s;
      }
    }
    asm volatile("s_wait_dscnt 0" ::: "memory");

    v8f acc[4];
    waveGemm<6>(tileG, 192, sW0, lane, acc);               // 192 -> 64
    lnRelu(acc, sPar + 0, sPar + 64, sPar + 128, lane);
    stageLds(acc, lane, tileG, 192);
    asm volatile("s_wait_dscnt 0" ::: "memory");

    waveGemm<2>(tileG, 192, sW1, lane, acc);               // 64 -> 64
    lnRelu(acc, sPar + 192, sPar + 256, sPar + 320, lane);
    stageLds(acc, lane, tileG, 192);
    asm volatile("s_wait_dscnt 0" ::: "memory");

    waveGemm<2>(tileG, 192, sW2, lane, acc);               // 64 -> 64
    lnRelu(acc, sPar + 384, sPar + 448, sPar + 512, lane);
    stageLds(acc, lane, tileG, 192);
    asm volatile("s_wait_dscnt 0" ::: "memory");

    waveGemm<2>(tileG, 192, sW3, lane, acc);               // 64 -> 64
    lnRelu(acc, sPar + 576, sPar + 640, sPar + 704, lane);

    // Residual (from gather tile cols 128..191) + store h.
    const int nl = lane & 15;
    const int mhi = (lane & 16) ? 8 : 0;
#pragma unroll
    for (int t = 0; t < 4; ++t)
#pragma unroll
      for (int r = 0; r < 8; ++r) {
        const int row = r + mhi;
        const int vv = n0 + row;
        if (vv < Nn) {
          const int n = nl + 16 * t;
          const float hold = (float)tileG[row * 192 + 128 + n];
          h[(size_t)vv * 64 + n] = (_Float16)(acc[t][r] + hold);
        }
      }
    asm volatile("s_wait_dscnt 0" ::: "memory");
  }
}

// ---------------------------------------------------------------------------
// Reductions and output
// ---------------------------------------------------------------------------
__global__ void hsum_kernel(const _Float16* __restrict__ h, float* __restrict__ hsum, int Nn) {
  const int tid = blockIdx.x * blockDim.x + threadIdx.x;
  const int col = tid & 63;
  const int rstride = (gridDim.x * blockDim.x) >> 6;
  float acc = 0.f;
  for (int v = tid >> 6; v < Nn; v += rstride) acc += (float)h[(size_t)v * 64 + col];
  atomicAdd(&hsum[col], acc);
}

__global__ void global_mlp_kernel(const float* __restrict__ hsum,
    const float* W0, const float* b0, const float* g0, const float* be0,
    const float* W1, const float* b1, const float* g1, const float* be1,
    const float* W2, const float* b2, const float* g2, const float* be2,
    const float* W3, const float* b3,
    float* __restrict__ sc, float* __restrict__ out_g1, float* __restrict__ out_g2) {
  __shared__ float xb[64];
  __shared__ float red[64];
  const int j = threadIdx.x;
  xb[j] = hsum[j]; __syncthreads();
  const float* Ws[3] = {W0, W1, W2};
  const float* bs[3] = {b0, b1, b2};
  const float* gs[3] = {g0, g1, g2};
  const float* es[3] = {be0, be1, be2};
#pragma unroll
  for (int l = 0; l < 3; ++l) {
    float o = bs[l][j];
    for (int k = 0; k < 64; ++k) o += xb[k] * Ws[l][k * 64 + j];
    red[j] = o; __syncthreads();
    for (int s = 32; s > 0; s >>= 1) { if (j < s) red[j] += red[j + s]; __syncthreads(); }
    float mu = red[0] * (1.f / 64.f); __syncthreads();
    red[j] = (o - mu) * (o - mu); __syncthreads();
    for (int s = 32; s > 0; s >>= 1) { if (j < s) red[j] += red[j + s]; __syncthreads(); }
    float var = red[0] * (1.f / 64.f); __syncthreads();
    float v = (o - mu) * rsqrtf(var + LN_EPS) * gs[l][j] + es[l][j];
    xb[j] = v > 0.f ? v : 0.f; __syncthreads();
  }
  if (j < 3) {
    float o = b3[j];
    for (int k = 0; k < 64; ++k) o += xb[k] * W3[k * 3 + j];
    if (j == 0) sc[0] = o;
    if (j == 1) *out_g1 = 1.f / (1.f + __expf(-o));
    if (j == 2) *out_g2 = 1.f / (1.f + __expf(-o));
  }
}

#define NB_RED 1024

__global__ void pmax_kernel(const float* __restrict__ ep, const float* __restrict__ sc,
                            float* __restrict__ part, int E) {
  __shared__ float red[256];
  const long total = (long)E + 1;
  const long stride = (long)gridDim.x * blockDim.x;
  float m = -3.4e38f;
  for (long i = (long)blockIdx.x * blockDim.x + threadIdx.x; i < total; i += stride)
    m = fmaxf(m, (i == 0) ? sc[0] : ep[i - 1]);
  red[threadIdx.x] = m; __syncthreads();
  for (int s = 128; s > 0; s >>= 1) {
    if (threadIdx.x < s) red[threadIdx.x] = fmaxf(red[threadIdx.x], red[threadIdx.x + s]);
    __syncthreads();
  }
  if (threadIdx.x == 0) part[blockIdx.x] = red[0];
}

__global__ void final_max_kernel(const float* __restrict__ part, float* __restrict__ sc, int nb) {
  __shared__ float red[256];
  float m = -3.4e38f;
  for (int i = threadIdx.x; i < nb; i += blockDim.x) m = fmaxf(m, part[i]);
  red[threadIdx.x] = m; __syncthreads();
  for (int s = 128; s > 0; s >>= 1) {
    if (threadIdx.x < s) red[threadIdx.x] = fmaxf(red[threadIdx.x], red[threadIdx.x + s]);
    __syncthreads();
  }
  if (threadIdx.x == 0) sc[1] = red[0];
}

__global__ void psum_kernel(const float* __restrict__ ep, const float* __restrict__ sc,
                            float* __restrict__ part, int E) {
  __shared__ float red[256];
  const float gm = sc[1];
  const long total = (long)E + 1;
  const long stride = (long)gridDim.x * blockDim.x;
  float s = 0.f;
  for (long i = (long)blockIdx.x * blockDim.x + threadIdx.x; i < total; i += stride)
    s += __expf(((i == 0) ? sc[0] : ep[i - 1]) - gm);
  red[threadIdx.x] = s; __syncthreads();
  for (int st = 128; st > 0; st >>= 1) {
    if (threadIdx.x < st) red[threadIdx.x] += red[threadIdx.x + st];
    __syncthreads();
  }
  if (threadIdx.x == 0) part[blockIdx.x] = red[0];
}

__global__ void final_sum_kernel(const float* __restrict__ part, float* __restrict__ sc, int nb) {
  __shared__ float red[256];
  float s = 0.f;
  for (int i = threadIdx.x; i < nb; i += blockDim.x) s += part[i];
  red[threadIdx.x] = s; __syncthreads();
  for (int st = 128; st > 0; st >>= 1) {
    if (threadIdx.x < st) red[threadIdx.x] += red[threadIdx.x + st];
    __syncthreads();
  }
  if (threadIdx.x == 0) sc[2] = logf(red[0]) + sc[1];
}

__global__ void finalize_kernel(const float* __restrict__ ep, const float* __restrict__ araw,
                                const float* __restrict__ sc, float* __restrict__ out, int E) {
  const float lz = sc[2];
  const long stride = (long)gridDim.x * blockDim.x;
  for (long i = (long)blockIdx.x * blockDim.x + threadIdx.x; i <= (long)E; i += stride)
    out[i] = ((i == 0) ? sc[0] : ep[i - 1]) - lz;
  float* aout = out + E + 2;
  for (long i = (long)blockIdx.x * blockDim.x + threadIdx.x; i < (long)E; i += stride)
    aout[i] = 1.f / (1.f + __expf(-araw[i]));
}

// ---------------------------------------------------------------------------
// Host orchestration
// ---------------------------------------------------------------------------
extern "C" void kernel_launch(void* const* d_in, const int* in_sizes, int n_in,
                              void* d_out, int out_size, void* d_ws, size_t ws_size,
                              hipStream_t stream) {
  const int Nn = in_sizes[0] / 3;
  const int E  = in_sizes[1] / 2;
  const float* x = (const float*)d_in[0];
  const int* startI = (const int*)d_in[1];
  const int* endI   = startI + E;
  auto F = [&](int i) { return (const float*)d_in[i]; };
  float* out = (float*)d_out;

  // Workspace carve (256B aligned).
  uintptr_t w = (uintptr_t)d_ws;
  auto take = [&](size_t bytes) {
    uintptr_t p = (w + 255) & ~(uintptr_t)255;
    w = p + bytes;
    return (void*)p;
  };
  _Float16* hbuf = (_Float16*)take((size_t)Nn * 64 * 2);
  float* mi   = (float*)take((size_t)Nn * 64 * 4);
  float* mo   = (float*)take((size_t)Nn * 64 * 4);
  float* eraw = (float*)take((size_t)E * 4);
  float* araw = (float*)take((size_t)E * 4);
  float* praw = (float*)take((size_t)E * 4);
  float* hsum = (float*)take(64 * 4);
  float* sc   = (float*)take(16 * 4);
  float* part = (float*)take(NB_RED * 4);
  _Float16* eWt0 = (_Float16*)take(64 * 128 * 2);
  _Float16* eWt1 = (_Float16*)take(64 * 64 * 2);
  _Float16* eWt2 = (_Float16*)take(64 * 64 * 2);
  _Float16* eW3  = (_Float16*)take(64 * 2);
  _Float16* nWt0 = (_Float16*)take(64 * 192 * 2);
  _Float16* nWt1 = (_Float16*)take(64 * 64 * 2);
  _Float16* nWt2 = (_Float16*)take(64 * 64 * 2);
  _Float16* nWt3 = (_Float16*)take(64 * 64 * 2);
  _Float16* pWt0 = (_Float16*)take(64 * 128 * 2);
  _Float16* pWt1 = (_Float16*)take(64 * 64 * 2);
  _Float16* pWt2 = (_Float16*)take(64 * 64 * 2);
  _Float16* pW3  = (_Float16*)take(64 * 2);

  // Weight prep: transpose + fp32 -> f16.
  transposeW_kernel<<<32, 256, 0, stream>>>(F(6),  eWt0, 128, 64);
  transposeW_kernel<<<16, 256, 0, stream>>>(F(10), eWt1, 64, 64);
  transposeW_kernel<<<16, 256, 0, stream>>>(F(14), eWt2, 64, 64);
  transposeW_kernel<<<1, 64, 0, stream>>>(F(18), eW3, 64, 1);
  transposeW_kernel<<<48, 256, 0, stream>>>(F(20), nWt0, 192, 64);
  transposeW_kernel<<<16, 256, 0, stream>>>(F(24), nWt1, 64, 64);
  transposeW_kernel<<<16, 256, 0, stream>>>(F(28), nWt2, 64, 64);
  transposeW_kernel<<<16, 256, 0, stream>>>(F(32), nWt3, 64, 64);
  transposeW_kernel<<<32, 256, 0, stream>>>(F(36), pWt0, 128, 64);
  transposeW_kernel<<<16, 256, 0, stream>>>(F(40), pWt1, 64, 64);
  transposeW_kernel<<<16, 256, 0, stream>>>(F(44), pWt2, 64, 64);
  transposeW_kernel<<<1, 64, 0, stream>>>(F(48), pW3, 64, 1);

  // Input MLP.
  input_mlp_kernel<<<Nn, 64, 0, stream>>>(x, F(2), F(3), F(4), F(5), hbuf, Nn);

  EdgeParams EP{eWt0, eWt1, eWt2, eW3,
                F(7), F(8), F(9), F(11), F(12), F(13), F(15), F(16), F(17), F(19)};
  EdgeParams PP{pWt0, pWt1, pWt2, pW3,
                F(37), F(38), F(39), F(41), F(42), F(43), F(45), F(46), F(47), F(49)};
  NodeParams NP;
  NP.Wt0 = nWt0; NP.Wt1 = nWt1; NP.Wt2 = nWt2; NP.Wt3 = nWt3;
  NP.b[0] = F(21); NP.g[0] = F(22); NP.be[0] = F(23);
  NP.b[1] = F(25); NP.g[1] = F(26); NP.be[1] = F(27);
  NP.b[2] = F(29); NP.g[2] = F(30); NP.be[2] = F(31);
  NP.b[3] = F(33); NP.g[3] = F(34); NP.be[3] = F(35);

  const size_t edgeSmem = (size_t)(64 * 128 + 64 * 64 * 2 + 64) * 2 +
                          (9 * 64 + 8) * 4 + (size_t)8 * 16 * 128 * 2;
  const size_t nodeSmem = (size_t)(64 * 192 + 3 * 64 * 64) * 2 +
                          (12 * 64) * 4 + (size_t)8 * 16 * 192 * 2;

  for (int it = 0; it < 3; ++it) {
    edge_mlp_kernel<<<1024, 256, edgeSmem, stream>>>(hbuf, startI, endI, EP, eraw, E);
    hipMemsetAsync(mi, 0, (size_t)Nn * 64 * 4, stream);
    hipMemsetAsync(mo, 0, (size_t)Nn * 64 * 4, stream);
    scatter_kernel<<<2048, 256, 0, stream>>>(hbuf, startI, endI, eraw, mi, mo, E);
    node_mlp_kernel<<<512, 256, nodeSmem, stream>>>(hbuf, mi, mo, NP, Nn);
  }

  edge_mlp_kernel<<<1024, 256, edgeSmem, stream>>>(hbuf, startI, endI, EP, araw, E);
  edge_mlp_kernel<<<1024, 256, edgeSmem, stream>>>(hbuf, startI, endI, PP, praw, E);

  hipMemsetAsync(hsum, 0, 64 * 4, stream);
  hsum_kernel<<<512, 256, 0, stream>>>(hbuf, hsum, Nn);
  global_mlp_kernel<<<1, 64, 0, stream>>>(hsum,
      F(50), F(51), F(52), F(53), F(54), F(55), F(56), F(57),
      F(58), F(59), F(60), F(61), F(62), F(63),
      sc, out + (E + 1), out + (2 * E + 2));

  pmax_kernel<<<NB_RED, 256, 0, stream>>>(praw, sc, part, E);
  final_max_kernel<<<1, 256, 0, stream>>>(part, sc, NB_RED);
  psum_kernel<<<NB_RED, 256, 0, stream>>>(praw, sc, part, E);
  final_sum_kernel<<<1, 256, 0, stream>>>(part, sc, NB_RED);
  finalize_kernel<<<NB_RED, 256, 0, stream>>>(praw, araw, sc, out, E);
}
